// MultiHeadAttention_6055903888045
// MI455X (gfx1250) — matrix operand order repaired and checked by reading, not compiled, not run on hardware
//
#include <hip/hip_runtime.h>

typedef _Float16 h16 __attribute__((ext_vector_type(16)));
typedef _Float16 h8v __attribute__((ext_vector_type(8)));
typedef _Float16 h4v __attribute__((ext_vector_type(4)));
typedef float    f8v __attribute__((ext_vector_type(8)));
typedef float    f4v __attribute__((ext_vector_type(4)));

#define D_MODEL 4096
#define D_HEAD  128
#define NHEADS  32

// ---- LDS geometry (units: _Float16 elements unless *_B = bytes) ----
#define WT_STRIDE 136   // W^T row stride (128 + 8 pad) -> conflict-free, 16B aligned
#define MS_STRIDE 136   // main row-major stride
#define MT_STRIDE 40    // main transposed stride (32 + 8 pad)
#define PT_STRIDE 40    // P chunk stride
#define AT_STRIDE 40    // attn stride
#define SC_STRIDE 33    // score stride (floats)

#define WT_BYTES (128 * WT_STRIDE * 2)  // 34816
#define MAINS_B  (32  * MS_STRIDE * 2)  // 8704
#define MAINT_B  (128 * MT_STRIDE * 2)  // 10240
#define PT_B     (32  * PT_STRIDE * 2)  // 2560
#define AT_B     (32  * AT_STRIDE * 2)  // 2560
#define SC_B     (32  * SC_STRIDE * 4)  // 4224
#define WAVE_B   (MAINS_B + MAINT_B + PT_B + AT_B + SC_B)   // 28288
#define SMEM_TOTAL (WT_BYTES + 8 * WAVE_B)                  // 261120 (<= 320KB WGP)

static __device__ __forceinline__ f8v wmma_f16(h16 a, h16 b, f8v c) {
  // D = A(16x32 f16) * B(32x16 f16) + C(16x16 f32)
  return __builtin_amdgcn_wmma_f32_16x16x32_f16(false, a, false, b, (short)0, c,
                                                false, false);
}

// A-fragment: lane holds A[M=lane&15][kb..kb+7] and A[M][kb+16..kb+23], kb = (lane>>4)*8
static __device__ __forceinline__ h16 ld_afrag(const _Float16* row, int kb) {
  h8v lo = *(const h8v*)(row + kb);
  h8v hi = *(const h8v*)(row + kb + 16);
  h16 r;
#pragma unroll
  for (int i = 0; i < 8; ++i) { r[i] = lo[i]; r[i + 8] = hi[i]; }
  return r;
}

// B-fragment: lane holds K kb..kb+7 and kb+16..kb+23 of column N=lane&15, kb = (lane>>4)*8 (p points at chunk base + (lane>>4)*16)
static __device__ __forceinline__ h16 ld_bfrag(const _Float16* p) {
  const int hb = ((int)(threadIdx.x & 31) >> 4) * 8;
  h8v lo = *(const h8v*)(p - hb);
  h8v hi = *(const h8v*)(p - hb + 16);
  h16 r;
#pragma unroll
  for (int i = 0; i < 8; ++i) { r[i] = lo[i]; r[i + 8] = hi[i]; }
  return r;
}

__global__ void __launch_bounds__(256, 1)
mha_wmma_kernel(const float* __restrict__ o1, const float* __restrict__ o2,
                const float* __restrict__ o3, const float* __restrict__ mainp,
                const float* __restrict__ Wp, float* __restrict__ outp, int Btot) {
  extern __shared__ char smem[];
  const int tid  = threadIdx.x;
  const int wave = tid >> 5;
  const int lane = tid & 31;
  const int lm   = lane & 15;
  const int hf   = lane >> 4;

  _Float16* WtS = (_Float16*)smem;
  char* wb = smem + WT_BYTES + wave * WAVE_B;
  _Float16* mainS  = (_Float16*)wb;
  _Float16* mainT  = (_Float16*)(wb + MAINS_B);
  _Float16* pTile  = (_Float16*)(wb + MAINS_B + MAINT_B);
  _Float16* attnS  = (_Float16*)(wb + MAINS_B + MAINT_B + PT_B);
  float*    scoreS = (float*)  (wb + MAINS_B + MAINT_B + PT_B + AT_B);

  // ---- stage W^T as f16 once per block (coalesced reads; padded rows) ----
  for (int i = tid; i < D_HEAD * D_HEAD; i += blockDim.x) {
    int d = i >> 7, e = i & 127;
    WtS[e * WT_STRIDE + d] = (_Float16)Wp[i];
  }
  __syncthreads();

  f8v zf = {0.f, 0.f, 0.f, 0.f, 0.f, 0.f, 0.f, 0.f};
  const int bstride = (int)gridDim.x * 8;

  for (int b = (int)blockIdx.x * 8 + wave; b < Btot; b += bstride) {
    const size_t base = (size_t)b * D_MODEL;

    // ---- stage main_b: row-major f16 + transposed f16 (float4 coalesced) ----
    const f4v* m4 = (const f4v*)(mainp + base);
#pragma unroll 4
    for (int g = 0; g < 32; ++g) {
      f4v v = m4[g * 32 + lane];            // row g, cols lane*4..lane*4+3
      int d = lane * 4;
      h4v hv;
#pragma unroll
      for (int t = 0; t < 4; ++t) hv[t] = (_Float16)v[t];
      *(h4v*)(mainS + g * MS_STRIDE + d) = hv;
#pragma unroll
      for (int t = 0; t < 4; ++t) mainT[(d + t) * MT_STRIDE + g] = hv[t];
    }

    // L2 prefetch of next batch row (global_prefetch_b8)
    {
      int nb = b + bstride;
      if (nb < Btot) __builtin_prefetch(mainp + (size_t)nb * D_MODEL + lane * 128, 0, 1);
    }

    // ---- mean A-fragments straight from global (each element read once) ----
    h16 aM[2][4];
#pragma unroll
    for (int m = 0; m < 2; ++m) {
      const int h = m * 16 + lm;
      const f4v* r1 = (const f4v*)(o1 + base + h * D_HEAD);
      const f4v* r2 = (const f4v*)(o2 + base + h * D_HEAD);
      const f4v* r3 = (const f4v*)(o3 + base + h * D_HEAD);
#pragma unroll
      for (int k = 0; k < 4; ++k) {
        const int i0 = k * 8 + hf * 2;   // float4 index of K = k*32 + hf*8
        h16 r;
#pragma unroll
        for (int q = 0; q < 4; ++q) {    // q 0,1 -> K+0..7 ; q 2,3 -> K+16..23
          int idx = i0 + (q & 1) + ((q >> 1) << 2);
          f4v s = r1[idx] + r2[idx] + r3[idx];
#pragma unroll
          for (int t = 0; t < 4; ++t) r[q * 4 + t] = (_Float16)(s[t] * (1.0f / 3.0f));
        }
        aM[m][k] = r;
      }
    }

    // ---- fused P = mean@W  and  score += P@main^T, e in chunks of 32 ----
    f8v sAcc[2][2];
    sAcc[0][0] = zf; sAcc[0][1] = zf; sAcc[1][0] = zf; sAcc[1][1] = zf;
#pragma unroll
    for (int j = 0; j < 4; ++j) {
#pragma unroll
      for (int nn = 0; nn < 2; ++nn) {
        const _Float16* wrow = WtS + (j * 32 + nn * 16 + lm) * WT_STRIDE;
        f8v p0 = zf, p1 = zf;
#pragma unroll
        for (int k = 0; k < 4; ++k) {
          h16 bw = ld_bfrag(wrow + k * 32 + hf * 16);
          p0 = wmma_f16(aM[0][k], bw, p0);
          p1 = wmma_f16(aM[1][k], bw, p1);
        }
        // C-layout -> LDS (f16) so it can be re-read in A-layout
#pragma unroll
        for (int v = 0; v < 8; ++v) {
          int r0 = v + hf * 8;
          pTile[r0 * PT_STRIDE + nn * 16 + lm]        = (_Float16)p0[v];
          pTile[(r0 + 16) * PT_STRIDE + nn * 16 + lm] = (_Float16)p1[v];
        }
      }
      // score accumulation for this 32-wide e-chunk
      h16 aS0 = ld_afrag(pTile + lm * PT_STRIDE,        hf * 8);
      h16 aS1 = ld_afrag(pTile + (16 + lm) * PT_STRIDE, hf * 8);
      const int e0 = j * 32 + hf * 16;
      h16 bS0 = ld_bfrag(mainS + lm * MS_STRIDE + e0);        // N = g = lm
      h16 bS1 = ld_bfrag(mainS + (16 + lm) * MS_STRIDE + e0); // N = g = 16+lm
      sAcc[0][0] = wmma_f16(aS0, bS0, sAcc[0][0]);
      sAcc[0][1] = wmma_f16(aS0, bS1, sAcc[0][1]);
      sAcc[1][0] = wmma_f16(aS1, bS0, sAcc[1][0]);
      sAcc[1][1] = wmma_f16(aS1, bS1, sAcc[1][1]);
    }

    // ---- softmax over g: spill 32x32 f32 score, one row per lane ----
#pragma unroll
    for (int m = 0; m < 2; ++m)
#pragma unroll
      for (int n = 0; n < 2; ++n)
#pragma unroll
        for (int v = 0; v < 8; ++v)
          scoreS[(m * 16 + v + hf * 8) * SC_STRIDE + n * 16 + lm] = sAcc[m][n][v];

    float buf[32];
    float mx = -3.402823466e38f;
#pragma unroll
    for (int c = 0; c < 32; ++c) {
      buf[c] = scoreS[lane * SC_STRIDE + c];
      mx = fmaxf(mx, buf[c]);
    }
    float sum = 0.f;
#pragma unroll
    for (int c = 0; c < 32; ++c) {
      float e = __expf(buf[c] - mx);
      buf[c] = e;
      sum += e;
    }
    float inv = 1.0f / sum;
#pragma unroll
    for (int c = 0; c < 32; ++c)
      attnS[lane * AT_STRIDE + c] = (_Float16)(buf[c] * inv);

    // ---- context = attn @ main (K=32, one WMMA per tile), stream to global ----
    h16 aC0 = ld_afrag(attnS + lm * AT_STRIDE,        hf * 8);
    h16 aC1 = ld_afrag(attnS + (16 + lm) * AT_STRIDE, hf * 8);
    float* outRow = outp + base;
#pragma unroll
    for (int n = 0; n < 8; ++n) {
      int d = n * 16 + lm;
      h16 bC = ld_bfrag(mainT + d * MT_STRIDE + hf * 16);  // column d of main
      f8v c0 = wmma_f16(aC0, bC, zf);
      f8v c1 = wmma_f16(aC1, bC, zf);
#pragma unroll
      for (int v = 0; v < 8; ++v) {
        int h0 = v + hf * 8;
        outRow[h0 * D_HEAD + d]        = c0[v];
        outRow[(h0 + 16) * D_HEAD + d] = c1[v];
      }
    }
  }
}

extern "C" void kernel_launch(void* const* d_in, const int* in_sizes, int n_in,
                              void* d_out, int out_size, void* d_ws, size_t ws_size,
                              hipStream_t stream) {
  const float* o1    = (const float*)d_in[0];
  const float* o2    = (const float*)d_in[1];
  const float* o3    = (const float*)d_in[2];
  const float* mainp = (const float*)d_in[3];
  const float* Wp    = (const float*)d_in[4];
  float* outp        = (float*)d_out;

  int Btot = in_sizes[3] / D_MODEL;   // 16384 for the reference shapes
  if (Btot <= 0) return;

  int nblocks = (Btot + 7) / 8;       // 8 batch rows (one per wave) per block-round
  if (nblocks > 512) nblocks = 512;   // grid-stride loop covers the rest

  mha_wmma_kernel<<<nblocks, 256, SMEM_TOTAL, stream>>>(o1, o2, o3, mainp, Wp,
                                                        outp, Btot);
}
